// Q_s2v_45105746542765
// MI455X (gfx1250) — matrix operand order repaired and checked by reading, not compiled, not run on hardware
//
#include <hip/hip_runtime.h>

// S2V GNN on MI455X (gfx1250, wave32).
//   - Edge-weight branch collapsed algebraically: relu(w*W4)=w*relu(W4) (w>=0)
//     -> per-node scalar sw[n] + rank-1 epilogue term. Computed once for all T.
//   - Layer GEMM (N x 128)(128 x 128) + rank-2 bias + relu fused, bf16 WMMA
//     (v_wmma_f32_16x16x32_bf16), f32 accumulation.
//   - Head fused: gterm[g] = sum_j relu((pooled@W6)[g,j])*W5[j];
//     out[n] = gterm[batch[n]] + sum_j relu((mu@W7)[n,j])*W5[128+j]
//     (WMMA + LDS ds_add_f32 row reduction).

typedef __attribute__((ext_vector_type(16))) __bf16 v16bf;
typedef __attribute__((ext_vector_type(8)))  float  v8f;

#define NN 50000
#define NE 800000
#define PD 128
#define TL 4
#define NG 8

// -------- per-node scalar edge-weight segment sum: sw[dst] += weight[e] -----
__global__ void k_sw(const float* __restrict__ w, const int* __restrict__ dst,
                     float* __restrict__ sw) {
  int e = blockIdx.x * blockDim.x + threadIdx.x;
  if (e < NE) atomicAdd(&sw[dst[e]], w[e]);
}

// -------- convert W2 (T*128*128) and W7 (128*128) to bf16 -------------------
__global__ void k_conv(const float* __restrict__ W2, const float* __restrict__ W7,
                       __bf16* __restrict__ W2bf, __bf16* __restrict__ W7bf) {
  int i = blockIdx.x * blockDim.x + threadIdx.x;
  if (i < TL * PD * PD) W2bf[i] = (__bf16)W2[i];
  else if (i < (TL + 1) * PD * PD) {
    int j = i - TL * PD * PD;
    W7bf[j] = (__bf16)W7[j];
  }
}

// -------- v3[t][j] = sum_k relu(W4[t][k]) * W3[t][k][j] ---------------------
__global__ void k_v3(const float* __restrict__ W3, const float* __restrict__ W4,
                     float* __restrict__ v3) {
  int i = blockIdx.x * blockDim.x + threadIdx.x;   // 512 total
  int t = i >> 7, j = i & 127;
  float s = 0.f;
  for (int k = 0; k < PD; ++k)
    s = fmaf(fmaxf(W4[t * PD + k], 0.f), W3[t * PD * PD + k * PD + j], s);
  v3[i] = s;
}

// -------- agg[dst] += mu[src] (float4 per thread, L2-resident atomics) ------
__global__ void k_scatter(const float* __restrict__ mu, const int* __restrict__ src,
                          const int* __restrict__ dst, float* __restrict__ agg) {
  int gid = blockIdx.x * blockDim.x + threadIdx.x;
  int e = gid >> 5;
  if (e >= NE) return;
  int c = (gid & 31) << 2;
  int s = src[e], d = dst[e];
  const float4 v = *(const float4*)(mu + (size_t)s * PD + c);
  float* p = agg + (size_t)d * PD + c;
  atomicAdd(p + 0, v.x); atomicAdd(p + 1, v.y);
  atomicAdd(p + 2, v.z); atomicAdd(p + 3, v.w);
}

// -------- fused layer: mu = relu(agg @ W2bf + x (x) W1t + sw (x) v3t) -------
// block = 128 thr (4 waves) = one 16-node tile; wave w owns col tiles {2w,2w+1}
__global__ void k_layer(const float* __restrict__ agg, const __bf16* __restrict__ W2t,
                        const float* __restrict__ x, const float* __restrict__ sw,
                        const float* __restrict__ W1t, const float* __restrict__ v3t,
                        float* __restrict__ mu) {
  int tid = threadIdx.x;
  int lane = tid & 31, wave = tid >> 5;
  int half = lane >> 4, mrow = lane & 15;
  int nb = blockIdx.x * 16;

  v8f c0 = {}, c1 = {};
#pragma unroll
  for (int kb = 0; kb < 4; ++kb) {
    // A tile 16x32: row = mrow, K elems {0..7,16..23} (+8 for upper lane half)
    v16bf a;
    const float* ar = agg + (size_t)(nb + mrow) * PD + kb * 32 + half * 8;
#pragma unroll
    for (int i = 0; i < 8; ++i) a[i] = (__bf16)ar[i];
#pragma unroll
    for (int i = 0; i < 8; ++i) a[8 + i] = (__bf16)ar[16 + i];
    // B tile 32x16: col = ct*16+mrow, K elems {0..7,16..23} (+8 for upper lane half)
    v16bf b0, b1;
#pragma unroll
    for (int i = 0; i < 8; ++i) {
      b0[i] = W2t[(size_t)(kb * 32 + half * 8 + i) * PD + (2 * wave + 0) * 16 + mrow];
      b0[8 + i] = W2t[(size_t)(kb * 32 + 16 + half * 8 + i) * PD + (2 * wave + 0) * 16 + mrow];
      b1[i] = W2t[(size_t)(kb * 32 + half * 8 + i) * PD + (2 * wave + 1) * 16 + mrow];
      b1[8 + i] = W2t[(size_t)(kb * 32 + 16 + half * 8 + i) * PD + (2 * wave + 1) * 16 + mrow];
    }
    c0 = __builtin_amdgcn_wmma_f32_16x16x32_bf16(false, a, false, b0, (short)0, c0, false, false);
    c1 = __builtin_amdgcn_wmma_f32_16x16x32_bf16(false, a, false, b1, (short)0, c1, false, false);
  }

  float xr[8], sr[8];
#pragma unroll
  for (int r = 0; r < 8; ++r) {
    int node = nb + r + 8 * half;
    xr[r] = x[node]; sr[r] = sw[node];
  }
#pragma unroll
  for (int cti = 0; cti < 2; ++cti) {
    int ncol = (2 * wave + cti) * 16 + mrow;
    float h1 = W1t[ncol], vv = v3t[ncol];
#pragma unroll
    for (int r = 0; r < 8; ++r) {
      int node = nb + r + 8 * half;
      float v = (cti == 0 ? c0[r] : c1[r]);
      v = fmaf(xr[r], h1, fmaf(sr[r], vv, v));
      mu[(size_t)node * PD + ncol] = fmaxf(v, 0.f);
    }
  }
}

// -------- pooled[g] += sum mu[n] over sorted batch (segmented, few atomics) -
__global__ void k_pool(const float* __restrict__ mu, const int* __restrict__ batch,
                       float* __restrict__ pooled) {
  int j = threadIdx.x;                       // column
  int n0 = blockIdx.x * 64, n1 = n0 + 64;
  if (n1 > NN) n1 = NN;
  float acc = 0.f;
  int cur = batch[n0];
  for (int n = n0; n < n1; ++n) {
    int b = batch[n];
    if (b != cur) { atomicAdd(&pooled[cur * PD + j], acc); acc = 0.f; cur = b; }
    acc += mu[(size_t)n * PD + j];
  }
  atomicAdd(&pooled[cur * PD + j], acc);
}

// -------- gterm[g] = sum_j relu((pooled@W6)[g,j]) * W5[j] -------------------
__global__ void k_head_pool(const float* __restrict__ pooled, const float* __restrict__ W6,
                            const float* __restrict__ W5, float* __restrict__ gterm) {
  __shared__ float red[PD];
  int g = blockIdx.x, j = threadIdx.x;
  float v = 0.f;
  for (int k = 0; k < PD; ++k) v = fmaf(pooled[g * PD + k], W6[k * PD + j], v);
  red[j] = fmaxf(v, 0.f) * W5[j];
  __syncthreads();
  for (int s = 64; s > 0; s >>= 1) { if (j < s) red[j] += red[j + s]; __syncthreads(); }
  if (j == 0) gterm[g] = red[0];
}

// -------- out[n] = gterm[batch[n]] + sum_j relu((mu@W7)[n,j]) * W5[128+j] ---
__global__ void k_final(const float* __restrict__ mu, const __bf16* __restrict__ W7bf,
                        const float* __restrict__ W5, const float* __restrict__ gterm,
                        const int* __restrict__ batch, float* __restrict__ out) {
  __shared__ float acc[16];
  int tid = threadIdx.x;
  int lane = tid & 31, wave = tid >> 5;
  int half = lane >> 4, mrow = lane & 15;
  int nb = blockIdx.x * 16;
  if (tid < 16) acc[tid] = 0.f;
  __syncthreads();

  v8f c0 = {}, c1 = {};
#pragma unroll
  for (int kb = 0; kb < 4; ++kb) {
    v16bf a;
    const float* ar = mu + (size_t)(nb + mrow) * PD + kb * 32 + half * 8;
#pragma unroll
    for (int i = 0; i < 8; ++i) a[i] = (__bf16)ar[i];
#pragma unroll
    for (int i = 0; i < 8; ++i) a[8 + i] = (__bf16)ar[16 + i];
    v16bf b0, b1;
#pragma unroll
    for (int i = 0; i < 8; ++i) {
      b0[i] = W7bf[(size_t)(kb * 32 + half * 8 + i) * PD + (2 * wave + 0) * 16 + mrow];
      b0[8 + i] = W7bf[(size_t)(kb * 32 + 16 + half * 8 + i) * PD + (2 * wave + 0) * 16 + mrow];
      b1[i] = W7bf[(size_t)(kb * 32 + half * 8 + i) * PD + (2 * wave + 1) * 16 + mrow];
      b1[8 + i] = W7bf[(size_t)(kb * 32 + 16 + half * 8 + i) * PD + (2 * wave + 1) * 16 + mrow];
    }
    c0 = __builtin_amdgcn_wmma_f32_16x16x32_bf16(false, a, false, b0, (short)0, c0, false, false);
    c1 = __builtin_amdgcn_wmma_f32_16x16x32_bf16(false, a, false, b1, (short)0, c1, false, false);
  }

  const float* w5b = W5 + PD;
#pragma unroll
  for (int cti = 0; cti < 2; ++cti) {
    int ncol = (2 * wave + cti) * 16 + mrow;
    float w5v = w5b[ncol];
#pragma unroll
    for (int r = 0; r < 8; ++r) {
      float v = (cti == 0 ? c0[r] : c1[r]);
      atomicAdd(&acc[r + 8 * half], fmaxf(v, 0.f) * w5v);   // ds_add_f32
    }
  }
  __syncthreads();
  if (tid < 16) {
    int node = nb + tid;
    out[node] = acc[tid] + gterm[batch[node]];
  }
}

extern "C" void kernel_launch(void* const* d_in, const int* in_sizes, int n_in,
                              void* d_out, int out_size, void* d_ws, size_t ws_size,
                              hipStream_t stream) {
  const float* x      = (const float*)d_in[0];
  const float* weight = (const float*)d_in[1];
  // d_in[2] = mu0 (all zeros) — never materialized; layer 0 skips the scatter.
  const int*   eidx   = (const int*)d_in[3];
  const int*   batch  = (const int*)d_in[4];
  const float* W1     = (const float*)d_in[5];
  const float* W2     = (const float*)d_in[6];
  const float* W3     = (const float*)d_in[7];
  const float* W4     = (const float*)d_in[8];
  const float* W5     = (const float*)d_in[9];
  const float* W6     = (const float*)d_in[10];
  const float* W7     = (const float*)d_in[11];
  float* out = (float*)d_out;
  const int* src = eidx;
  const int* dst = eidx + NE;

  // workspace carve-up (~51.6 MB)
  char* ws = (char*)d_ws;
  size_t off = 0;
  auto carve = [&](size_t bytes) -> char* {
    char* p = ws + off;
    off += (bytes + 255) & ~(size_t)255;
    return p;
  };
  float*  mu     = (float*)carve((size_t)NN * PD * 4);
  float*  agg    = (float*)carve((size_t)NN * PD * 4);
  float*  sw     = (float*)carve((size_t)NN * 4);
  float*  pooled = (float*)carve((size_t)NG * PD * 4);
  float*  gterm  = (float*)carve((size_t)NG * 4);
  float*  v3     = (float*)carve((size_t)TL * PD * 4);
  __bf16* W2bf   = (__bf16*)carve((size_t)TL * PD * PD * 2);
  __bf16* W7bf   = (__bf16*)carve((size_t)PD * PD * 2);

  hipMemsetAsync(sw, 0, (size_t)NN * 4, stream);
  hipMemsetAsync(pooled, 0, (size_t)NG * PD * 4, stream);

  k_sw<<<(NE + 255) / 256, 256, 0, stream>>>(weight, dst, sw);
  k_conv<<<((TL + 1) * PD * PD + 255) / 256, 256, 0, stream>>>(W2, W7, W2bf, W7bf);
  k_v3<<<TL, PD, 0, stream>>>(W3, W4, v3);

  for (int t = 0; t < TL; ++t) {
    hipMemsetAsync(agg, 0, (size_t)NN * PD * 4, stream);
    if (t > 0)
      k_scatter<<<(NE * 32 + 127) / 128, 128, 0, stream>>>(mu, src, dst, agg);
    k_layer<<<NN / 16, 128, 0, stream>>>(agg, W2bf + (size_t)t * PD * PD, x, sw,
                                         W1 + t * PD, v3 + t * PD, mu);
  }

  k_pool<<<(NN + 63) / 64, PD, 0, stream>>>(mu, batch, pooled);
  k_head_pool<<<NG, PD, 0, stream>>>(pooled, W6, W5, gterm);
  k_final<<<NN / 16, 128, 0, stream>>>(mu, W7bf, W5, gterm, batch, out);
}
